// LinearKAN_1357209666064
// MI455X (gfx1250) — matrix operand order repaired and checked by reading, not compiled, not run on hardware
//
#include <hip/hip_runtime.h>

// ---------------------------------------------------------------------------
// LinearKAN: out[b,o] = sum_{ch=0..8} sum_i F_ch(x[b,i]) * W_ch[o,i]
//   F_ch = exp(-((x-g_ch)*1.75)^2) for ch<8,  F_8 = silu(x)
//   W_ch = c[ch]*w_s for ch<8,                W_8 = w_b
// One bf16 WMMA GEMM: M=16384, N=768, K=9*768.
// RBF trick: s=1.75^2, h=4/7 -> s*h^2 = 1, 2*s*h = 3.5. Center on grid pt 3:
//   f_a = P * Q^(a-3) * e^{-(a-3)^2},  P = exp(-s*tc^2), Q = exp(3.5*tc),
//   tc = x + 2/7.  => 2 exps + 1 rcp for all 8 channels.
// ---------------------------------------------------------------------------

typedef __bf16 v16bf __attribute__((ext_vector_type(16)));
typedef __bf16 v2bf  __attribute__((ext_vector_type(2)));
typedef float  v8f   __attribute__((ext_vector_type(8)));

#define BATCH   16384
#define IN_F    768
#define OUT_F   768
#define NCH     9
#define KCH     32                  // i-elements per K chunk (WMMA K=32)
#define NCHUNK  (IN_F / KCH)        // 24
#define BM      128
#define BN      192                 // 4x feature redundancy
#define BUFQ    (NCH * BM * 4)      // uint4 per LDS feature buffer

#if defined(__has_builtin)
#if __has_builtin(__builtin_amdgcn_cvt_pk_bf16_f32)
#define HAVE_PK_BF16 1
#endif
#endif

__device__ __forceinline__ unsigned short f2bf(float f) {
  unsigned int u = __float_as_uint(f);
  unsigned int r = u + 0x7FFFu + ((u >> 16) & 1u);
  return (unsigned short)(r >> 16);
}

__device__ __forceinline__ unsigned pk2(float a, float b) {
#ifdef HAVE_PK_BF16
  v2bf v = __builtin_amdgcn_cvt_pk_bf16_f32(a, b);
  union { v2bf v; unsigned u; } c; c.v = v; return c.u;
#else
  return (unsigned)f2bf(a) | ((unsigned)f2bf(b) << 16);
#endif
}

__device__ __forceinline__ v16bf frag_from(uint4 lo, uint4 hi) {
  union { uint4 q[2]; v16bf v; } u;
  u.q[0] = lo; u.q[1] = hi;
  return u.v;
}

// all 9 feature channels of one x value (3 exps + 2 rcps total)
__device__ __forceinline__ void features(float x, float f[NCH]) {
  const float LOG2E = 1.4426950408889634f;
  const float K1  = 0.36787944117144233f;    // e^-1
  const float K4  = 0.018315638888734179f;   // e^-4
  const float K9  = 1.2340980408667956e-4f;  // e^-9
  const float K16 = 1.1253517471925912e-7f;  // e^-16
  const float xc = fminf(fmaxf(x, -4.2f), 4.2f);
  const float tc = xc + (2.0f / 7.0f);       // x - g3, g3 = -2/7
  const float P = __builtin_amdgcn_exp2f(-(3.0625f * LOG2E) * tc * tc);
  const float Q = __builtin_amdgcn_exp2f((3.5f * LOG2E) * tc);
  const float R = __builtin_amdgcn_rcpf(Q);
  f[3] = P;
  float u = Q;
  f[4] = P * u * K1;  u *= Q;
  f[5] = P * u * K4;  u *= Q;
  f[6] = P * u * K9;  u *= Q;
  f[7] = P * u * K16;
  float v = R;
  f[2] = P * v * K1;  v *= R;
  f[1] = P * v * K4;  v *= R;
  f[0] = P * v * K9;
  const float sg = __builtin_amdgcn_rcpf(1.0f + __builtin_amdgcn_exp2f(-x * LOG2E));
  f[8] = x * sg;                             // silu
}

// ---------------------------------------------------------------------------
// Kernel 1: pack weights to bf16, fragment-friendly layout:
//   Wpk[ch][chunk][o][k] (k = i % 32), ch<8: c*w_s, ch==8: w_b
// ---------------------------------------------------------------------------
__global__ void __launch_bounds__(256) kan_pack_weights(
    const float* __restrict__ w_b, const float* __restrict__ w_s,
    const float* __restrict__ c, unsigned short* __restrict__ Wpk) {
  const long total = (long)NCH * NCHUNK * OUT_F * KCH;
  long idx = (long)blockIdx.x * 256 + threadIdx.x;
  if (idx >= total) return;
  int kk = (int)(idx & (KCH - 1));
  long t = idx >> 5;
  int o = (int)(t % OUT_F); t /= OUT_F;
  int chunk = (int)(t % NCHUNK); t /= NCHUNK;
  int ch = (int)t;
  int i = chunk * KCH + kk;
  float v;
  if (ch < 8) v = c[((long)ch * OUT_F + o) * IN_F + i] * w_s[(long)o * IN_F + i];
  else        v = w_b[(long)o * IN_F + i];
  Wpk[idx] = f2bf(v);
}

// ---------------------------------------------------------------------------
// Kernel 2: fused feature-expansion + bf16 WMMA GEMM, software-pipelined.
// Block = 256 threads (8 wave32), tile 128(M) x 192(N).
// Wave tile 32x96 -> 2x6 accumulators of 16x16 f32 (96 VGPRs).
// LDS (147 KB) caps occupancy at 2 WGs/WGP = 4 waves/SIMD; tell the RA to
// target 2 waves/EU so it can keep several B fragments in flight instead of
// serializing each WMMA pair behind s_wait_loadcnt 0x0.
// ---------------------------------------------------------------------------
__global__ void __launch_bounds__(256, 2) kan_wmma_gemm(
    const float* __restrict__ X, const unsigned short* __restrict__ Wpk,
    float* __restrict__ Out) {
  // two bf16 feature buffers: [buf][ch][m][k], row = 32 bf16 = 64 B = 4 uint4
  __shared__ uint4 ldsF[2 * BUFQ];   // 147,456 B

  const int tid  = threadIdx.x;
  const int lane = tid & 31;
  const int wid  = tid >> 5;
  const int wm   = wid & 3;              // m offset = 32*wm
  const int wn   = wid >> 2;             // n offset = 96*wn
  const int bBase = blockIdx.x * BM;
  const int nBase = blockIdx.y * BN;

  v8f acc[2][6];
#pragma unroll
  for (int tm = 0; tm < 2; ++tm)
#pragma unroll
    for (int tn = 0; tn < 6; ++tn)
      acc[tm][tn] = v8f{0.f, 0.f, 0.f, 0.f, 0.f, 0.f, 0.f, 0.f};

  const int col4  = (tid & 7) * 4;       // i offset within chunk (float4)
  const int rbase = tid >> 3;            // 0..31, rows rbase + 32*r

  // ---- feature staging: [128 x 32] x-tile -> 9 bf16 channels in LDS ----
  auto stage = [&](int chunk, int buf) {
    const int i0 = chunk * KCH;
    uint2* fdst = (uint2*)&ldsF[buf * BUFQ];
#pragma unroll
    for (int r = 0; r < 4; ++r) {
      const int m = rbase + 32 * r;
      const float4 xv =
          *(const float4*)&X[(long)(bBase + m) * IN_F + i0 + col4];
      if (chunk + 2 < NCHUNK)
        __builtin_prefetch(&X[(long)(bBase + m) * IN_F + i0 + 2 * KCH + col4],
                           0, 0);
      const float xs[4] = {xv.x, xv.y, xv.z, xv.w};
      float fv[4][NCH];
#pragma unroll
      for (int e = 0; e < 4; ++e) features(xs[e], fv[e]);
#pragma unroll
      for (int ch = 0; ch < NCH; ++ch) {
        uint2 p;
        p.x = pk2(fv[0][ch], fv[1][ch]);
        p.y = pk2(fv[2][ch], fv[3][ch]);
        fdst[(ch * BM + m) * 8 + (col4 >> 2)] = p;
      }
    }
  };

  stage(0, 0);
  __syncthreads();

  const int loSel = (lane < 16) ? 0 : 1;     // A frag: K0..7/16..23 vs 8..15/24..31
  const int mrow0 = wm * 32 + (lane & 15);
  // B frag base: column (lane&15) of this wave's N panel; lanes>=16 take K8..15/24..31
  const unsigned short* wbase =
      Wpk + (long)(nBase + wn * 96 + (lane & 15)) * KCH + ((lane < 16) ? 0 : 8);

  for (int chunk = 0; chunk < NCHUNK; ++chunk) {
    // issue next chunk's x loads + feature math before this chunk's WMMAs
    if (chunk + 1 < NCHUNK) stage(chunk + 1, (chunk + 1) & 1);

    const uint4* fsrc = &ldsF[(chunk & 1) * BUFQ];
    const unsigned short* wch = wbase + (long)chunk * (OUT_F * KCH);
#pragma unroll
    for (int ch = 0; ch < NCH; ++ch) {
      v16bf a[2];
#pragma unroll
      for (int tm = 0; tm < 2; ++tm) {
        const uint4* row = &fsrc[(ch * BM + (mrow0 + tm * 16)) * 4];
        a[tm] = frag_from(row[loSel], row[loSel + 2]);
      }
      // batch all 6 B fragments -> one clause, progressive loadcnt waits
      v16bf b[6];
#pragma unroll
      for (int tn = 0; tn < 6; ++tn) {
        const unsigned short* p =
            wch + (long)ch * (NCHUNK * OUT_F * KCH) + tn * (16 * KCH);
        b[tn] = frag_from(*(const uint4*)p, *(const uint4*)(p + 16));
      }
#pragma unroll
      for (int tn = 0; tn < 6; ++tn) {
        acc[0][tn] = __builtin_amdgcn_wmma_f32_16x16x32_bf16(
            false, a[0], false, b[tn], (short)0, acc[0][tn], false, false);
        acc[1][tn] = __builtin_amdgcn_wmma_f32_16x16x32_bf16(
            false, a[1], false, b[tn], (short)0, acc[1][tn], false, false);
      }
    }
    __syncthreads();   // all reads of buf[chunk&1] done before rewrite (2 ahead)
  }

  // ---- epilogue: D layout = VGPR r: lanes0-15 M=r, lanes16-31 M=8+r ----
#pragma unroll
  for (int tm = 0; tm < 2; ++tm)
#pragma unroll
    for (int tn = 0; tn < 6; ++tn) {
      const int o  = nBase + wn * 96 + tn * 16 + (lane & 15);
      const int mb = bBase + wm * 32 + tm * 16 + ((lane < 16) ? 0 : 8);
#pragma unroll
      for (int r = 0; r < 8; ++r)
        Out[(long)(mb + r) * OUT_F + o] = acc[tm][tn][r];
    }
}

// ---------------------------------------------------------------------------
extern "C" void kernel_launch(void* const* d_in, const int* in_sizes, int n_in,
                              void* d_out, int out_size, void* d_ws, size_t ws_size,
                              hipStream_t stream) {
  const float* x   = (const float*)d_in[0];
  const float* w_b = (const float*)d_in[1];
  const float* w_s = (const float*)d_in[2];
  const float* c   = (const float*)d_in[3];
  unsigned short* Wpk = (unsigned short*)d_ws;   // 9*24*768*32*2 B = 10.6 MB

  const long total = (long)NCH * NCHUNK * OUT_F * KCH;
  kan_pack_weights<<<(int)((total + 255) / 256), 256, 0, stream>>>(w_b, w_s, c, Wpk);

  dim3 grid(BATCH / BM, OUT_F / BN);
  kan_wmma_gemm<<<grid, 256, 0, stream>>>(x, Wpk, (float*)d_out);
}
